// VectorQuantizer_62216896250294
// MI455X (gfx1250) — matrix operand order repaired and checked by reading, not compiled, not run on hardware
//
#include <hip/hip_runtime.h>
#include <hip/hip_bf16.h>

typedef __attribute__((ext_vector_type(16))) __bf16 v16bf;
typedef __attribute__((ext_vector_type(8)))  float  v8f;
typedef __attribute__((ext_vector_type(4)))  float  v4f;
typedef __attribute__((ext_vector_type(4)))  int    v4i;
typedef __attribute__((address_space(1))) v4i gv4i;   // global-space v4i
typedef __attribute__((address_space(3))) v4i lv4i;   // LDS-space v4i

#define D_DIM 256
#define K_DIM 4096
#define NQ    16777216          // 64*1024*256 quantized elements
#define OFF_DICT   (NQ)
#define OFF_COMMIT (NQ + 1)
#define OFF_IDX    (NQ + 2)

#define TPS          4                       // code tiles per LDS stage
#define TILE_ELEMS   (8 * 32 * 16)           // bf16 elems per code tile (8 KB)
#define STAGE_ELEMS  (TPS * TILE_ELEMS)      // 16384 bf16 = 32 KB
#define NSTAGES      (K_DIM / 16 / TPS)      // 64

#if __has_builtin(__builtin_amdgcn_global_load_async_to_lds_b128)
#define HAS_ASYNC_LDS 1
#else
#define HAS_ASYNC_LDS 0
#endif

// copy one 16-byte chunk global -> LDS (async if the gfx1250 path is available)
__device__ __forceinline__ void copy16_g2l(const __bf16* __restrict__ g,
                                           __bf16* __restrict__ l) {
#if HAS_ASYNC_LDS
    __builtin_amdgcn_global_load_async_to_lds_b128((gv4i*)g, (lv4i*)l, 0, 0);
#else
    *(v4i*)l = *(const v4i*)g;   // vmem load + ds_store fallback
#endif
}

__device__ __forceinline__ void wait_async_copies() {
#if __has_builtin(__builtin_amdgcn_s_wait_asynccnt)
    __builtin_amdgcn_s_wait_asynccnt(0);
#elif HAS_ASYNC_LDS
    asm volatile("s_wait_asynccnt 0x0" ::: "memory");
#endif
}

// whole-stage cooperative copy: 32 KB, 256 threads x 8 x 16B, fully coalesced
__device__ __forceinline__ void stage_copy(__bf16* __restrict__ dst,
                                           const __bf16* __restrict__ src, int t) {
#pragma unroll
    for (int i = 0; i < 8; ++i) {
        int e = (i * 256 + t) * 8;           // 16B chunk -> 8 bf16 elems
        copy16_g2l(src + e, dst + e);
    }
}

__device__ __forceinline__ v16bf lds_b(const __bf16* buf, int tile, int kc, int lane) {
    return *(const v16bf*)(buf + ((tile * 8 + kc) * 32 + lane) * 16);
}

// ---------------- prep kernels ----------------

__global__ __launch_bounds__(256) void vq_vnorm_kernel(const float* __restrict__ vec,
                                                       float* __restrict__ vnorm) {
    int k = blockIdx.x * 256 + threadIdx.x;
    float s = 0.0f;
    for (int d = 0; d < D_DIM; ++d) {
        float v = vec[(size_t)d * K_DIM + k];
        s += v * v;
    }
    vnorm[k] = s;
}

__global__ __launch_bounds__(256) void vq_vt_kernel(const float* __restrict__ vec,
                                                    float* __restrict__ vt) {
    int t = blockIdx.x * 256 + threadIdx.x;   // t over K*D
    int d = t & (D_DIM - 1);
    int k = t >> 8;
    vt[t] = vec[(size_t)d * K_DIM + k];
}

// vb: bf16 codebook in WMMA-B lane layout.
// flat index = ((ct*8 + kc)*32 + lane)*16 + e
// lane = n + 16*h holds column n of code-tile ct, K-slice kc: k = kc*32 + h*8 + (e&7) + 16*(e>>3)
__global__ __launch_bounds__(256) void vq_vb_kernel(const float* __restrict__ vec,
                                                    __bf16* __restrict__ vb) {
    int t = blockIdx.x * 256 + threadIdx.x;   // t over K*D
    int e    = t & 15;
    int lane = (t >> 4) & 31;
    int kc   = (t >> 9) & 7;
    int ct   = t >> 12;
    int n = lane & 15;
    int h = lane >> 4;
    int k    = kc * 32 + h * 8 + (e & 7) + (e >> 3) * 16;          // inner (embedding) dim
    int code = ct * 16 + n;
    vb[t] = (__bf16)vec[(size_t)k * K_DIM + code];
}

// ---------------- fused GEMM + argmin (LDS-staged, register double-buffered B) ----
__global__ __launch_bounds__(256) void vq_argmin_kernel(const float* __restrict__ x,
                                                        const __bf16* __restrict__ vb,
                                                        const float* __restrict__ vnorm,
                                                        int* __restrict__ idxOut) {
    __shared__ __bf16 Bs[2][STAGE_ELEMS];     // 2 x 32 KB

    const int t    = threadIdx.x;
    const int lane = t & 31;
    const int wave = t >> 5;
    const int r0   = (blockIdx.x * 8 + wave) * 16;
    const int m    = lane & 15;
    const int h    = lane >> 4;

    // A-chunks (16x32 bf16 each) built once from f32 x; reused for all code tiles.
    // A layout: lane m+16h, elems 0..7 -> K = kc*32 + h*8 + e ; elems 8..15 -> +16
    v16bf a[8];
    const float* rowp = x + (size_t)(r0 + m) * D_DIM;
#pragma unroll
    for (int kc = 0; kc < 8; ++kc) {
        const float* p = rowp + kc * 32 + h * 8;
        v4f f0 = *(const v4f*)(p);
        v4f f1 = *(const v4f*)(p + 4);
        v4f f2 = *(const v4f*)(p + 16);
        v4f f3 = *(const v4f*)(p + 20);
        v16bf av;
#pragma unroll
        for (int e = 0; e < 4; ++e) {
            av[e]      = (__bf16)f0[e];
            av[e + 4]  = (__bf16)f1[e];
            av[e + 8]  = (__bf16)f2[e];
            av[e + 12] = (__bf16)f3[e];
        }
        a[kc] = av;
    }

    float best[8];
    int   bidx[8];
#pragma unroll
    for (int j = 0; j < 8; ++j) { best[j] = 3.4e38f; bidx[j] = 0; }

    // prologue: stage 0 into buffer 0
    stage_copy(&Bs[0][0], vb, t);

    for (int s = 0; s < NSTAGES; ++s) {
        wait_async_copies();
        __syncthreads();                      // stage s resident; all reads of other buf done
        if (s + 1 < NSTAGES)
            stage_copy(&Bs[(s + 1) & 1][0], vb + (size_t)(s + 1) * STAGE_ELEMS, t);

        const __bf16* buf = &Bs[s & 1][0];

        // hoist per-stage vnorm loads so they clause ahead of compute
        float vns[TPS];
#pragma unroll
        for (int tile = 0; tile < TPS; ++tile)
            vns[tile] = vnorm[(s * TPS + tile) * 16 + m];

        // register double buffer for B operands: preload tile 0
        v16bf b0[8], b1[8];
#pragma unroll
        for (int kc = 0; kc < 8; ++kc) b0[kc] = lds_b(buf, 0, kc, lane);

#pragma unroll
        for (int tile = 0; tile < TPS; ++tile) {
            v16bf (&cur)[8] = (tile & 1) ? b1 : b0;
            v16bf (&nxt)[8] = (tile & 1) ? b0 : b1;
            if (tile + 1 < TPS) {
#pragma unroll
                for (int kc = 0; kc < 8; ++kc) nxt[kc] = lds_b(buf, tile + 1, kc, lane);
            }
            // two accumulators break the WMMA->WMMA C-chain for matrix-pipe ILP
            v8f acc0 = {}, acc1 = {};
#pragma unroll
            for (int kc = 0; kc < 8; kc += 2) {
                acc0 = __builtin_amdgcn_wmma_f32_16x16x32_bf16(false, a[kc], false, cur[kc],
                                                               (short)0, acc0, false, false);
                acc1 = __builtin_amdgcn_wmma_f32_16x16x32_bf16(false, a[kc + 1], false, cur[kc + 1],
                                                               (short)0, acc1, false, false);
            }
            // C layout: lanes 0-15 hold N=lane, rows M=j; lanes 16-31: N=lane-16, M=j+8
            const int code = (s * TPS + tile) * 16 + m;
            const float vn = vns[tile];
#pragma unroll
            for (int j = 0; j < 8; ++j) {
                float sc = vn - 2.0f * (acc0[j] + acc1[j]);  // ||v||^2 - 2 x.v
                if (sc < best[j]) { best[j] = sc; bidx[j] = code; }
            }
        }
    }

    // butterfly min+argmin across the 16 lanes of each half (xor 1,2,4,8 stays in half)
#pragma unroll
    for (int j = 0; j < 8; ++j) {
#pragma unroll
        for (int off = 1; off < 16; off <<= 1) {
            float os = __shfl_xor(best[j], off, 32);
            int   oi = __shfl_xor(bidx[j], off, 32);
            if (os < best[j] || (os == best[j] && oi < bidx[j])) { best[j] = os; bidx[j] = oi; }
        }
    }
    if (m == 0) {
#pragma unroll
        for (int j = 0; j < 8; ++j)
            idxOut[r0 + h * 8 + j] = bidx[j];
    }
}

// ---------------- epilogue ----------------

__global__ void vq_zero_kernel(float* __restrict__ out) {
    if (threadIdx.x < 2) out[NQ + threadIdx.x] = 0.0f;
}

__global__ __launch_bounds__(256) void vq_gather_kernel(const float* __restrict__ x,
                                                        const float* __restrict__ vt,
                                                        const int* __restrict__ idxIn,
                                                        float* __restrict__ out,
                                                        float inv_total) {
    __shared__ float wsum[8];
    const int r = blockIdx.x;
    const int d = threadIdx.x;
    const int idx = idxIn[r];
    const size_t o = (size_t)r * D_DIM + d;
    float q  = vt[(size_t)idx * D_DIM + d];
    float df = x[o] - q;
    out[o] = q;                       // straight-through forward value == q
    float s = df * df;
#pragma unroll
    for (int off = 16; off; off >>= 1) s += __shfl_xor(s, off, 32);
    if ((d & 31) == 0) wsum[d >> 5] = s;
    __syncthreads();
    if (d == 0) {
        float tt = 0.0f;
#pragma unroll
        for (int w = 0; w < 8; ++w) tt += wsum[w];
        tt *= inv_total;
        atomicAdd(out + OFF_DICT, tt);     // dictionary loss
        atomicAdd(out + OFF_COMMIT, tt);   // commitment loss (same forward value)
        out[OFF_IDX + r] = (float)idx;     // indices, as output dtype (float)
    }
}

extern "C" void kernel_launch(void* const* d_in, const int* in_sizes, int n_in,
                              void* d_out, int out_size, void* d_ws, size_t ws_size,
                              hipStream_t stream) {
    const float* x   = (const float*)d_in[0];   // (64,1024,256) f32
    const float* vec = (const float*)d_in[1];   // (256,4096) f32
    float* out = (float*)d_out;

    const int N = in_sizes[0] / D_DIM;          // 65536 rows

    // workspace carve-up (~6.3 MB)
    char* ws = (char*)d_ws;
    __bf16* vb    = (__bf16*)(ws);                                  // 2 MB
    float*  vt    = (float*)(ws + (2u << 20));                      // 4 MB
    float*  vnorm = (float*)(ws + (6u << 20));                      // 16 KB
    int*    idxI  = (int*)(ws + (6u << 20) + (1u << 16));           // 256 KB

    vq_vnorm_kernel<<<K_DIM / 256, 256, 0, stream>>>(vec, vnorm);
    vq_vt_kernel<<<(K_DIM * D_DIM) / 256, 256, 0, stream>>>(vec, vt);
    vq_vb_kernel<<<(K_DIM * D_DIM) / 256, 256, 0, stream>>>(vec, vb);

    vq_argmin_kernel<<<N / (16 * 8), 256, 0, stream>>>(x, vb, vnorm, idxI);

    vq_zero_kernel<<<1, 64, 0, stream>>>(out);
    vq_gather_kernel<<<N, 256, 0, stream>>>(x, vt, idxI, out,
                                            1.0f / (float)((size_t)N * D_DIM));
}
